// HiSACKT_NoSkillClustering_11046655885799
// MI455X (gfx1250) — matrix operand order repaired and checked by reading, not compiled, not run on hardware
//
#include <hip/hip_runtime.h>
#include <hip/hip_bf16.h>

// ---------------------------------------------------------------------------
// HiSACKT forward for MI455X (gfx1250, wave32).
// All GEMM-shaped math runs on v_wmma_f32_16x16x32_bf16 (f32 accum).
// GEMM uses LDS-staged 128x128 tiles (bf16 panels) + global_prefetch_b8.
// Workspace layout: 10 buffers of B*N*D f32 each (~168 MB total).
// ---------------------------------------------------------------------------

#define B_   16
#define N_   512
#define D_   512
#define H_   8
#define DH_  64
#define NQ_  8192
#define LW_  25
#define NE_  ((long)B_ * N_ * D_)   // 4,194,304 elements per activation buffer

#define GT_M 128
#define GT_N 128
#define GT_K 32

typedef __attribute__((ext_vector_type(16))) __bf16 v16bf;
typedef __attribute__((ext_vector_type(8)))  float  v8f;

__device__ __forceinline__ v8f zero_v8f() {
  v8f z;
#pragma unroll
  for (int i = 0; i < 8; ++i) z[i] = 0.0f;
  return z;
}

// A-fragment (16x32 bf16): lane 0-15 = rows, holds K 0..15;
// lane 16-31 same rows, K 16..31.  rowPtr = &A[row*ld + k0] (f32).
__device__ __forceinline__ v16bf fragA_rowmajor(const float* __restrict__ rowPtr,
                                                int laneHi) {
  const float* p = rowPtr + laneHi * 16;
  v16bf f;
#pragma unroll
  for (int i = 0; i < 8; ++i) f[i] = (__bf16)p[i];
#pragma unroll
  for (int i = 0; i < 8; ++i) f[8 + i] = (__bf16)p[8 + i];
  return f;
}

// B-fragment (32x16 bf16): lane = column n; lanes 0-15 hold K 0..15,
// lanes 16-31 hold K 16..31, sequential in K. rowPtr = &W[col*ld + k0] (f32).
__device__ __forceinline__ v16bf fragB_from_rows(const float* __restrict__ rowPtr,
                                                 int laneHi) {
  const float* p = rowPtr + laneHi * 16;
  v16bf f;
#pragma unroll
  for (int i = 0; i < 16; ++i) f[i] = (__bf16)p[i];
  return f;
}

__device__ __forceinline__ v8f wmma_bf16(v16bf a, v16bf b, v8f c) {
  return __builtin_amdgcn_wmma_f32_16x16x32_bf16(false, a, false, b,
                                                 (short)0, c, false, false);
}

// ---------------------------------------------------------------------------
// Generic batched GEMM:  C[b,i,o] = act( sum_k A[b,i,k] * W[b,o,k] + bias[o] )
// A row stride = K, W row stride = K, C row stride = outD.
// Block = 256 threads (8 waves) computing a 128x128 tile via LDS-staged
// bf16 panels. Wave (wr,wc) owns rows [wr*32,+32), cols [wc*64,+64).
// ---------------------------------------------------------------------------
__global__ __launch_bounds__(256)
void gemm_bf16_kernel(const float* __restrict__ A, const float* __restrict__ W,
                      const float* __restrict__ bias, float* __restrict__ C,
                      int rows, int K, int outD, int act,
                      long sA, long sW, long sC)
{
  __shared__ __bf16 Abuf[GT_M][GT_K];   // [row][k]
  __shared__ __bf16 Bbuf[GT_N][GT_K];   // [col][k]

  const int tid    = threadIdx.x;
  const int lane   = tid & 31;
  const int wave   = tid >> 5;
  const int laneHi = lane >> 4;
  const int lane16 = lane & 15;
  const int wr = wave >> 1;                  // 0..3 : 32-row group
  const int wc = wave & 1;                   // 0..1 : 64-col group
  const int rowBase = blockIdx.x * GT_M;
  const int colBase = blockIdx.y * GT_N;
  const int bz = blockIdx.z;
  A += (long)bz * sA;  W += (long)bz * sW;  C += (long)bz * sC;

  // Stager: thread t fills row (t>>1), 16-element half (t&1) of both panels.
  const int ldRow  = tid >> 1;
  const int ldHalf = (tid & 1) * 16;
  int aRow = rowBase + ldRow;  if (aRow >= rows) aRow = rows - 1;
  int wCol = colBase + ldRow;  if (wCol >= outD) wCol = outD - 1;
  const float* aPtr = A + (long)aRow * K + ldHalf;
  const float* wPtr = W + (long)wCol * K + ldHalf;

  v8f acc[2][4];
#pragma unroll
  for (int sr = 0; sr < 2; ++sr)
#pragma unroll
    for (int sc = 0; sc < 4; ++sc) acc[sr][sc] = zero_v8f();

  for (int k = 0; k < K; k += GT_K) {
    if (k + GT_K < K) {                       // prefetch next panels
      __builtin_prefetch(aPtr + GT_K, 0, 1);  // -> global_prefetch_b8
      __builtin_prefetch(wPtr + GT_K, 0, 1);
    }
    // ---- stage f32 -> bf16 into LDS ----
    {
      const float4* a4 = (const float4*)aPtr;
      const float4* w4 = (const float4*)wPtr;
      float4 av[4], wv[4];
#pragma unroll
      for (int i = 0; i < 4; ++i) { av[i] = a4[i]; wv[i] = w4[i]; }
      __bf16* as = &Abuf[ldRow][ldHalf];
      __bf16* ws = &Bbuf[ldRow][ldHalf];
#pragma unroll
      for (int i = 0; i < 4; ++i) {
        as[i*4+0] = (__bf16)av[i].x;  as[i*4+1] = (__bf16)av[i].y;
        as[i*4+2] = (__bf16)av[i].z;  as[i*4+3] = (__bf16)av[i].w;
        ws[i*4+0] = (__bf16)wv[i].x;  ws[i*4+1] = (__bf16)wv[i].y;
        ws[i*4+2] = (__bf16)wv[i].z;  ws[i*4+3] = (__bf16)wv[i].w;
      }
    }
    __syncthreads();

    // ---- 8 WMMAs per wave per k-step from LDS ----
    v16bf aF[2];
#pragma unroll
    for (int sr = 0; sr < 2; ++sr) {
      const __bf16* p = &Abuf[wr * 32 + sr * 16 + lane16][laneHi * 16];
#pragma unroll
      for (int i = 0; i < 8; ++i) { aF[sr][i] = p[i]; aF[sr][8 + i] = p[8 + i]; }
    }
#pragma unroll
    for (int sc = 0; sc < 4; ++sc) {
      const __bf16* p = &Bbuf[wc * 64 + sc * 16 + lane16][laneHi * 16];
      v16bf bF;
#pragma unroll
      for (int i = 0; i < 16; ++i) bF[i] = p[i];
      acc[0][sc] = wmma_bf16(aF[0], bF, acc[0][sc]);
      acc[1][sc] = wmma_bf16(aF[1], bF, acc[1][sc]);
    }
    __syncthreads();
    aPtr += GT_K;  wPtr += GT_K;
  }

  // ---- epilogue: bias + act + store ----
#pragma unroll
  for (int sc = 0; sc < 4; ++sc) {
    int col = colBase + wc * 64 + sc * 16 + lane16;
    if (col >= outD) continue;
    float bv = bias ? bias[col] : 0.0f;
#pragma unroll
    for (int sr = 0; sr < 2; ++sr) {
#pragma unroll
      for (int v = 0; v < 8; ++v) {
        int row = rowBase + wr * 32 + sr * 16 + v + 8 * laneHi;
        if (row < rows) {
          float x = acc[sr][sc][v] + bv;
          if (act == 1) x = x > 0.0f ? x : 0.0f;
          C[(long)row * outD + col] = x;
        }
      }
    }
  }
}

// ---------------------------------------------------------------------------
// Fused attention core. One wave per (batch b, 16-row q tile); loops heads.
// Q (b,n_q,D), K/V (b,n_k,D), head h in cols [h*64, h*64+64).
// scores -> mask -> softmax (LDS, f32) -> ctx via WMMA.
// If wMean != null, writes mean-over-heads probs (deterministic, no atomics).
// ---------------------------------------------------------------------------
__global__ __launch_bounds__(32)
void attn_core_kernel(const float* __restrict__ Qm, const float* __restrict__ Km,
                      const float* __restrict__ Vm, float* __restrict__ Cm,
                      float* __restrict__ wMean, int n_q, int n_k, int causal)
{
  __shared__ float P[16][516];      // score/prob tile (padded)
  __shared__ float WACC[16][512];   // head-mean prob accumulator

  const int lane   = threadIdx.x;
  const int laneHi = lane >> 4;
  const int lane16 = lane & 15;
  const int qBase  = blockIdx.x * 16;
  const int b      = blockIdx.z;
  const long qOff  = (long)b * n_q * D_;
  const long kOff  = (long)b * n_k * D_;

  int qrow = qBase + lane16;  if (qrow >= n_q) qrow = n_q - 1;
  const int nTiles = (n_k + 15) >> 4;
  const int kLimit = nTiles << 4;          // <= 512

  for (int h = 0; h < H_; ++h) {
    const float* Q = Qm + qOff + h * DH_;
    const float* K = Km + kOff + h * DH_;
    const float* V = Vm + kOff + h * DH_;
    const float* qPtr = Q + (long)qrow * D_;

    // ---- scores = (Qh . Khᵀ) / sqrt(dh) ----
    for (int nt = 0; nt < nTiles; ++nt) {
      int krow = nt * 16 + lane16;
      int krowC = krow < n_k ? krow : n_k - 1;
      const float* kPtr = K + (long)krowC * D_;
      v8f acc = zero_v8f();
#pragma unroll
      for (int ks = 0; ks < DH_; ks += 32) {
        v16bf a  = fragA_rowmajor(qPtr + ks, laneHi);
        v16bf bf = fragB_from_rows(kPtr + ks, laneHi);
        acc = wmma_bf16(a, bf, acc);
      }
      int keyIdx = nt * 16 + lane16;
#pragma unroll
      for (int v = 0; v < 8; ++v) {
        int m = v + 8 * laneHi;
        int qIdx = qBase + m;
        float s = acc[v] * 0.125f;            // 1/sqrt(64)
        if (keyIdx >= n_k || (causal && keyIdx > qIdx)) s = -1e30f;
        P[m][keyIdx] = s;
      }
    }
    __syncthreads();

    // ---- softmax: lanes 0..15 each own one query row ----
    if (lane < 16) {
      int qIdx = qBase + lane;
      if (qIdx >= n_q) {
        for (int k = 0; k < kLimit; ++k) P[lane][k] = 0.0f;
      } else {
        float mx = P[lane][0];
        for (int k = 1; k < kLimit; ++k) mx = fmaxf(mx, P[lane][k]);
        float sum = 0.0f;
        for (int k = 0; k < kLimit; ++k) {
          float e = __expf(P[lane][k] - mx);
          P[lane][k] = e;  sum += e;
        }
        float inv = 1.0f / sum;
        for (int k = 0; k < kLimit; ++k) P[lane][k] *= inv;
        if (wMean) {
          const float hs = 1.0f / (float)H_;
          if (h == 0) for (int k = 0; k < kLimit; ++k) WACC[lane][k]  = P[lane][k] * hs;
          else        for (int k = 0; k < kLimit; ++k) WACC[lane][k] += P[lane][k] * hs;
        }
      }
    }
    __syncthreads();

    // ---- ctx = P . Vh ----
#pragma unroll
    for (int dt = 0; dt < 4; ++dt) {
      int dcol = dt * 16 + lane16;
      v8f acc = zero_v8f();
      for (int ks = 0; ks < n_k; ks += 32) {
        v16bf a;
        {
          int kb = ks + laneHi * 16;
#pragma unroll
          for (int i = 0; i < 8; ++i) a[i]     = (__bf16)P[lane16][kb + i];
#pragma unroll
          for (int i = 0; i < 8; ++i) a[8 + i] = (__bf16)P[lane16][kb + 8 + i];
        }
        v16bf bf;
        {
          int kb = ks + laneHi * 16;
#pragma unroll
          for (int i = 0; i < 16; ++i) {
            int k = kb + i;
            bf[i] = (k < n_k) ? (__bf16)V[(long)k * D_ + dcol] : (__bf16)0.0f;
          }
        }
        acc = wmma_bf16(a, bf, acc);
      }
#pragma unroll
      for (int v = 0; v < 8; ++v) {
        int row = qBase + v + 8 * laneHi;
        if (row < n_q) Cm[qOff + (long)row * D_ + h * DH_ + dcol] = acc[v];
      }
    }
    __syncthreads();
  }

  if (wMean && lane < 16) {
    int qIdx = qBase + lane;
    if (qIdx < n_q) {
      float* wrow = wMean + ((long)b * n_q + qIdx) * n_k;
      for (int k = 0; k < n_k; ++k) wrow[k] = WACC[lane][k];
    }
  }
}

// ---------------------------------------------------------------------------
// Elementwise / reduction helpers
// ---------------------------------------------------------------------------
__global__ void gather_M_kernel(const float* __restrict__ Mtab,
                                const float* __restrict__ Ptab,
                                const int* __restrict__ q, const int* __restrict__ r,
                                float* __restrict__ out)
{
  long i = (long)blockIdx.x * blockDim.x + threadIdx.x;
  if (i >= NE_) return;
  int  d  = (int)(i % D_);
  long bn = i / D_;
  int  n  = (int)(bn % N_);
  int  x  = q[bn] + NQ_ * r[bn];
  out[i] = Mtab[(long)x * D_ + d] + Ptab[(long)n * D_ + d];
}

__global__ void gather_E_kernel(const float* __restrict__ Etab,
                                const int* __restrict__ qry, float* __restrict__ out)
{
  long i = (long)blockIdx.x * blockDim.x + threadIdx.x;
  if (i >= NE_) return;
  int  d  = (int)(i % D_);
  long bn = i / D_;
  out[i] = Etab[(long)qry[bn] * D_ + d];
}

__global__ void gather_skill_kernel(const float* __restrict__ Wsp,
                                    const float* __restrict__ bsp,
                                    const int* __restrict__ q, float* __restrict__ out)
{
  long i = (long)blockIdx.x * blockDim.x + threadIdx.x;
  if (i >= NE_) return;
  int  d  = (int)(i % D_);
  long bn = i / D_;
  out[i] = Wsp[(long)d * NQ_ + q[bn]] + bsp[d];   // skill_proj_W.T[q] + b
}

__global__ void zero_kernel(float* __restrict__ p, long n)
{
  long i = (long)blockIdx.x * blockDim.x + threadIdx.x;
  if (i < n) p[i] = 0.0f;
}

__global__ void add6_kernel(const float* a, const float* b, const float* c,
                            const float* d, const float* e, const float* f,
                            float* __restrict__ out, long n)
{
  long i = (long)blockIdx.x * blockDim.x + threadIdx.x;
  if (i < n) out[i] = a[i] + b[i] + c[i] + d[i] + e[i] + f[i];
}

// out[row] = LN(x1 + x2 + x3)  (x2/x3 nullable). One wave per row of D=512.
__global__ __launch_bounds__(32)
void ln_residual_kernel(const float* __restrict__ x1, const float* __restrict__ x2,
                        const float* __restrict__ x3, const float* __restrict__ g,
                        const float* __restrict__ bta, float* __restrict__ out)
{
  long row = blockIdx.x;
  const long base = row * D_;
  float vals[16];
  float s = 0.0f;
#pragma unroll
  for (int i = 0; i < 16; ++i) {
    int d = threadIdx.x + i * 32;
    float v = x1[base + d];
    if (x2) v += x2[base + d];
    if (x3) v += x3[base + d];
    vals[i] = v;  s += v;
  }
#pragma unroll
  for (int off = 16; off > 0; off >>= 1) s += __shfl_xor(s, off, 32);
  float mu = s * (1.0f / D_);
  float vs = 0.0f;
#pragma unroll
  for (int i = 0; i < 16; ++i) { float d = vals[i] - mu; vs += d * d; }
#pragma unroll
  for (int off = 16; off > 0; off >>= 1) vs += __shfl_xor(vs, off, 32);
  float inv = rsqrtf(vs * (1.0f / D_) + 1e-5f);
#pragma unroll
  for (int i = 0; i < 16; ++i) {
    int d = threadIdx.x + i * 32;
    out[base + d] = (vals[i] - mu) * inv * g[d] + bta[d];
  }
}

// p_out[row] = sigmoid(F[row] . predW + predB). One wave per row.
__global__ __launch_bounds__(32)
void pred_kernel(const float* __restrict__ F, const float* __restrict__ Wp,
                 const float* __restrict__ bp, float* __restrict__ out)
{
  long row = blockIdx.x;
  const long base = row * D_;
  float s = 0.0f;
#pragma unroll
  for (int i = 0; i < 16; ++i) {
    int d = threadIdx.x + i * 32;
    s += F[base + d] * Wp[d];
  }
#pragma unroll
  for (int off = 16; off > 0; off >>= 1) s += __shfl_xor(s, off, 32);
  if (threadIdx.x == 0) out[row] = 1.0f / (1.0f + __expf(-(s + bp[0])));
}

// ---------------------------------------------------------------------------
// Orchestration
// ---------------------------------------------------------------------------
extern "C" void kernel_launch(void* const* d_in, const int* in_sizes, int n_in,
                              void* d_out, int out_size, void* d_ws, size_t ws_size,
                              hipStream_t stream)
{
  (void)in_sizes; (void)n_in; (void)out_size; (void)ws_size;

  struct Att { const float *Wq,*Wk,*Wv,*bq,*bk,*bv,*Wo,*bo; };
  auto att = [&](int i) -> Att {
    return Att{ (const float*)d_in[i+0], (const float*)d_in[i+1],
                (const float*)d_in[i+2], (const float*)d_in[i+3],
                (const float*)d_in[i+4], (const float*)d_in[i+5],
                (const float*)d_in[i+6], (const float*)d_in[i+7] };
  };

  const float* Mtab = (const float*)d_in[0];
  const float* Etab = (const float*)d_in[1];
  const float* Ptab = (const float*)d_in[2];
  Att aBase  = att(3),  aLocal = att(11), aSkill = att(19);
  Att aGlob  = att(27), aBL    = att(35), aSG    = att(43);
  const float* ln1_g = (const float*)d_in[51];
  const float* ln1_b = (const float*)d_in[52];
  const float* spW   = (const float*)d_in[53];
  const float* spB   = (const float*)d_in[54];
  const float* seW   = (const float*)d_in[55];
  const float* seB   = (const float*)d_in[56];
  const float* fW1   = (const float*)d_in[57];
  const float* fB1   = (const float*)d_in[58];
  const float* fW2   = (const float*)d_in[59];
  const float* fB2   = (const float*)d_in[60];
  const float* ln2_g = (const float*)d_in[61];
  const float* ln2_b = (const float*)d_in[62];
  const float* prW   = (const float*)d_in[63];
  const float* prB   = (const float*)d_in[64];
  const int*   q     = (const int*)d_in[65];
  const int*   r     = (const int*)d_in[66];
  const int*   qry   = (const int*)d_in[67];

  float* out  = (float*)d_out;
  float* wOut = out + (long)B_ * N_;          // attn_w_base region

  float* ws = (float*)d_ws;
  auto buf = [&](int i) { return ws + (long)i * NE_; };
  float *b0 = buf(0), *b1 = buf(1), *b2 = buf(2), *b3 = buf(3), *b4 = buf(4),
        *b5 = buf(5), *b6 = buf(6), *b7 = buf(7), *b8 = buf(8), *b9 = buf(9);

  auto gemm = [&](const float* A, const float* W, const float* bias, float* C,
                  int rows, int K, int outD, int act, int batch,
                  long sA, long sW, long sC) {
    dim3 g((rows + GT_M - 1) / GT_M, (outD + GT_N - 1) / GT_N, batch);
    gemm_bf16_kernel<<<g, 256, 0, stream>>>(A, W, bias, C, rows, K, outD, act,
                                            sA, sW, sC);
  };
  auto attn = [&](const float* Q, const float* K, const float* V, float* C,
                  float* wm, int n_q, int n_k, int causal) {
    dim3 g((n_q + 15) / 16, 1, B_);
    attn_core_kernel<<<g, 32, 0, stream>>>(Q, K, V, C, wm, n_q, n_k, causal);
  };

  const int RN = B_ * N_;                       // 8192 rows
  const int eb = 256;
  const long eg = (NE_ + eb - 1) / eb;

  // 1) Embedding gathers: M -> b0, E -> b1
  gather_M_kernel<<<eg, eb, 0, stream>>>(Mtab, Ptab, q, r, b0);
  gather_E_kernel<<<eg, eb, 0, stream>>>(Etab, qry, b1);

  // 2) Base attention (Q from E, K/V from M, causal) + ln1 -> S_base = b7
  gemm(b1, aBase.Wq, aBase.bq, b2, RN, D_, D_, 0, 1, 0, 0, 0);
  gemm(b0, aBase.Wk, aBase.bk, b3, RN, D_, D_, 0, 1, 0, 0, 0);
  gemm(b0, aBase.Wv, aBase.bv, b4, RN, D_, D_, 0, 1, 0, 0, 0);
  attn(b2, b3, b4, b5, wOut, N_, N_, 1);
  gemm(b5, aBase.Wo, aBase.bo, b6, RN, D_, D_, 0, 1, 0, 0, 0);
  ln_residual_kernel<<<RN, 32, 0, stream>>>(b6, b0, b1, ln1_g, ln1_b, b7);

  // 3) Local attention on last LW rows -> S_local_full = b0 (zero elsewhere)
  const float* locIn = b7 + (long)(N_ - LW_) * D_;
  gemm(locIn, aLocal.Wq, aLocal.bq, b2, LW_, D_, D_, 0, B_, (long)N_*D_, 0, (long)LW_*D_);
  gemm(locIn, aLocal.Wk, aLocal.bk, b3, LW_, D_, D_, 0, B_, (long)N_*D_, 0, (long)LW_*D_);
  gemm(locIn, aLocal.Wv, aLocal.bv, b4, LW_, D_, D_, 0, B_, (long)N_*D_, 0, (long)LW_*D_);
  attn(b2, b3, b4, b5, nullptr, LW_, LW_, 1);
  zero_kernel<<<eg, eb, 0, stream>>>(b0, NE_);
  gemm(b5, aLocal.Wo, aLocal.bo, b0 + (long)(N_ - LW_) * D_,
       LW_, D_, D_, 0, B_, (long)LW_*D_, 0, (long)N_*D_);

  // 4) Skill branch -> S_skill = b1
  gather_skill_kernel<<<eg, eb, 0, stream>>>(spW, spB, q, b1);          // repr
  gemm(b7, b1, nullptr, b2, N_, D_, N_, 0, B_,
       (long)N_*D_, (long)N_*D_, (long)N_*N_);                          // bnm
  gemm(b2, seW, seB, b6, RN, N_, D_, 0, 1, 0, 0, 0);                    // -> (B,N,D)
  gemm(b6, aSkill.Wq, aSkill.bq, b2, RN, D_, D_, 0, 1, 0, 0, 0);
  gemm(b6, aSkill.Wk, aSkill.bk, b3, RN, D_, D_, 0, 1, 0, 0, 0);
  gemm(b6, aSkill.Wv, aSkill.bv, b4, RN, D_, D_, 0, 1, 0, 0, 0);
  attn(b2, b3, b4, b5, nullptr, N_, N_, 0);
  gemm(b5, aSkill.Wo, aSkill.bo, b1, RN, D_, D_, 0, 1, 0, 0, 0);

  // 5) Global self-attention (causal) -> S_global = b6
  gemm(b7, aGlob.Wq, aGlob.bq, b2, RN, D_, D_, 0, 1, 0, 0, 0);
  gemm(b7, aGlob.Wk, aGlob.bk, b3, RN, D_, D_, 0, 1, 0, 0, 0);
  gemm(b7, aGlob.Wv, aGlob.bv, b4, RN, D_, D_, 0, 1, 0, 0, 0);
  attn(b2, b3, b4, b5, nullptr, N_, N_, 1);
  gemm(b5, aGlob.Wo, aGlob.bo, b6, RN, D_, D_, 0, 1, 0, 0, 0);

  // 6) Cross base<-local -> S_bl = b8
  gemm(b7, aBL.Wq, aBL.bq, b2, RN, D_, D_, 0, 1, 0, 0, 0);
  gemm(b0, aBL.Wk, aBL.bk, b3, RN, D_, D_, 0, 1, 0, 0, 0);
  gemm(b0, aBL.Wv, aBL.bv, b4, RN, D_, D_, 0, 1, 0, 0, 0);
  attn(b2, b3, b4, b5, nullptr, N_, N_, 0);
  gemm(b5, aBL.Wo, aBL.bo, b8, RN, D_, D_, 0, 1, 0, 0, 0);

  // 7) Cross skill<-global -> S_sg = b9
  gemm(b1, aSG.Wq, aSG.bq, b2, RN, D_, D_, 0, 1, 0, 0, 0);
  gemm(b6, aSG.Wk, aSG.bk, b3, RN, D_, D_, 0, 1, 0, 0, 0);
  gemm(b6, aSG.Wv, aSG.bv, b4, RN, D_, D_, 0, 1, 0, 0, 0);
  attn(b2, b3, b4, b5, nullptr, N_, N_, 0);
  gemm(b5, aSG.Wo, aSG.bo, b9, RN, D_, D_, 0, 1, 0, 0, 0);

  // 8) S = sum of six branches -> b2
  add6_kernel<<<eg, eb, 0, stream>>>(b7, b0, b1, b6, b8, b9, b2, NE_);

  // 9) FFN + ln2 -> b5
  gemm(b2, fW1, fB1, b3, RN, D_, D_, 1, 1, 0, 0, 0);     // relu
  gemm(b3, fW2, fB2, b4, RN, D_, D_, 0, 1, 0, 0, 0);
  ln_residual_kernel<<<RN, 32, 0, stream>>>(b4, b2, nullptr, ln2_g, ln2_b, b5);

  // 10) Prediction head -> d_out[0 .. B*N)
  pred_kernel<<<RN, 32, 0, stream>>>(b5, prW, prB, out);
}
